// SelfDotProductAttnQKNorm_44727789421361
// MI455X (gfx1250) — matrix operand order repaired and checked by reading, not compiled, not run on hardware
//
#include <hip/hip_runtime.h>

// ---------------------------------------------------------------------------
// Problem constants (reference: B=4, H=16, S=2048, D=64, fp32)
// ---------------------------------------------------------------------------
#define BH    64          // B*H
#define SEQ   2048
#define DHEAD 64
#define WAVES 8           // waves per block in attention kernel (wave32)

typedef __attribute__((ext_vector_type(16))) _Float16 v16h;
typedef __attribute__((ext_vector_type(8)))  float    v8f;
typedef __attribute__((ext_vector_type(4)))  unsigned u32x4;

union AB { v16h v; u32x4 q[2]; };

static_assert(sizeof(v16h) == 32, "v16h must be 8 VGPRs");

// 16-lane (half-wave) max reduction: xor masks < 16 stay inside each half of
// the wave32, matching the WMMA C-layout (rows 0-7 in lanes 0-15, 8-15 above).
__device__ __forceinline__ float hmax16(float x) {
#pragma unroll
  for (int off = 1; off < 16; off <<= 1) x = fmaxf(x, __shfl_xor(x, off, 32));
  return x;
}

// ---------------------------------------------------------------------------
// Pass 1: RMSNorm rows of 64 fp32 -> f16, optional (gain * 1/sqrt(D)) folded in.
// ---------------------------------------------------------------------------
__global__ __launch_bounds__(256) void rmsnorm_f32_to_f16(
    const float* __restrict__ src, _Float16* __restrict__ dst,
    const float* __restrict__ gain, float extraScale) {
  const int lane = threadIdx.x & 31;
  const int wave = threadIdx.x >> 5;
  const size_t row = (size_t)blockIdx.x * 8 + wave;     // 8 waves per block
  const float2 v = ((const float2*)src)[row * 32 + lane];
  float ss = v.x * v.x + v.y * v.y;
#pragma unroll
  for (int off = 1; off < 32; off <<= 1) ss += __shfl_xor(ss, off, 32);
  float inv = rsqrtf(ss * (1.0f / 64.0f) + 1e-6f) * extraScale;
  if (gain) inv *= gain[0];
  union { _Float16 h[2]; unsigned u; } p;
  p.h[0] = (_Float16)(v.x * inv);
  p.h[1] = (_Float16)(v.y * inv);
  ((unsigned*)dst)[row * 32 + lane] = p.u;
}

// ---------------------------------------------------------------------------
// Pass 2: V fp32 [bh, s, d] -> f16 transposed [bh, d, s] via LDS tile so the
// P·V B-matrix loads become contiguous 16B vector loads along the key index.
// ---------------------------------------------------------------------------
__global__ __launch_bounds__(256) void v_f32_to_f16_T(
    const float* __restrict__ V, _Float16* __restrict__ vt) {
  __shared__ _Float16 tile[64][72];                     // pad vs bank conflicts
  const int bh = blockIdx.y;
  const int s0 = blockIdx.x * 64;
  const float* src = V + ((size_t)bh * SEQ + s0) * DHEAD;
#pragma unroll
  for (int i = 0; i < 16; ++i) {
    int idx = threadIdx.x + 256 * i;
    int sl = idx >> 6, d = idx & 63;                    // coalesced global read
    tile[d][sl] = (_Float16)src[idx];
  }
  __syncthreads();
  _Float16* dstb = vt + (size_t)bh * DHEAD * SEQ + s0;
#pragma unroll
  for (int i = 0; i < 16; ++i) {
    int idx = threadIdx.x + 256 * i;
    int d = idx >> 6, sl = idx & 63;                    // coalesced global write
    dstb[(size_t)d * SEQ + sl] = tile[d][sl];
  }
}

// ---------------------------------------------------------------------------
// Pass 3: flash attention. One wave32 per 16-row query tile, 32-key tiles,
// online softmax (row-sum done on the tensor core via P x ones), diagonal
// tile peeled so the main loop carries no mask/exec manipulation.
// ---------------------------------------------------------------------------
__global__ __launch_bounds__(256) void attn_flash_wmma(
    const _Float16* __restrict__ qn,   // [bh, s, d]  normalized Q (scale folded)
    const _Float16* __restrict__ kn,   // [bh, s, d]  normalized K
    const _Float16* __restrict__ vt,   // [bh, d, s]  V transposed
    float* __restrict__ out) {         // [bh, s, d]
  __shared__ __align__(16) _Float16 pbuf[WAVES][16][32];  // P staging (per wave)

  const int lane = threadIdx.x & 31;
  const int wave = threadIdx.x >> 5;
  const int bh   = blockIdx.y;
  const int q0   = (blockIdx.x * WAVES + wave) * 16;

  const int n    = lane & 15;          // col (B/C layouts) and row m (A layout)
  const int hi   = lane >> 4;          // which half of the wave
  const int base = hi * 8;             // A-layout octet base

  // ---- Q as two A-matrices (16x32 f16, contiguous layout: k = 16*hi + s)
  const _Float16* qrow = qn + ((size_t)bh * SEQ + q0 + n) * DHEAD;
  AB a_lo, a_hi;
  a_lo.q[0] = *(const u32x4*)(qrow + 16 * hi);         // d = 16hi..16hi+7
  a_lo.q[1] = *(const u32x4*)(qrow + 16 * hi + 8);     // d = 16hi+8..16hi+15
  a_hi.q[0] = *(const u32x4*)(qrow + 32 + 16 * hi);
  a_hi.q[1] = *(const u32x4*)(qrow + 32 + 16 * hi + 8);

  // ---- all-ones B-matrix: row-sum of P as a WMMA (P x 1 -> rowsums in C)
  AB onesb;
#pragma unroll
  for (int i = 0; i < 16; ++i) onesb.v[i] = (_Float16)1.0f;

  v8f o0 = {}, o1 = {}, o2 = {}, o3 = {};
  float mrow[8], lrow[8];
#pragma unroll
  for (int v = 0; v < 8; ++v) { mrow[v] = -3.0e38f; lrow[v] = 0.0f; }

  const _Float16* kbase = kn + (size_t)bh * SEQ * DHEAD;
  const _Float16* vbase = vt + (size_t)bh * DHEAD * SEQ;
  const int nT = (q0 + 47) >> 5;       // ceil((q0+16)/32) causal key tiles

  // Per-tile body; `domask` is always a compile-time constant at call sites so
  // the masked/unmasked versions are separate straight-line specializations.
  auto tile_step = [&](int k0, bool domask) {
    // ---- K^T as B-matrices (32x16): lane n = key column, halves cover the
    //      gemm-K (=d) range 16*hi..16*hi+15, contiguous in memory.
    const _Float16* krL = kbase + (size_t)(k0 + n) * DHEAD;
    const _Float16* krR = kbase + (size_t)(k0 + 16 + n) * DHEAD;
    __builtin_prefetch(krL + 32 * DHEAD, 0, 3);        // next key tile
    AB bLlo, bLhi, bRlo, bRhi;
    bLlo.q[0] = *(const u32x4*)(krL + 16 * hi);
    bLlo.q[1] = *(const u32x4*)(krL + 16 * hi + 8);
    bLhi.q[0] = *(const u32x4*)(krL + 32 + 16 * hi);
    bLhi.q[1] = *(const u32x4*)(krL + 32 + 16 * hi + 8);
    bRlo.q[0] = *(const u32x4*)(krR + 16 * hi);
    bRlo.q[1] = *(const u32x4*)(krR + 16 * hi + 8);
    bRhi.q[0] = *(const u32x4*)(krR + 32 + 16 * hi);
    bRhi.q[1] = *(const u32x4*)(krR + 32 + 16 * hi + 8);

    // ---- scores: two 16x16 f32 tiles, each = two K=32 WMMAs over d
    v8f sL = {}, sR = {};
    sL = __builtin_amdgcn_wmma_f32_16x16x32_f16(false, a_lo.v, false, bLlo.v,
                                                (short)0, sL, false, false);
    sL = __builtin_amdgcn_wmma_f32_16x16x32_f16(false, a_hi.v, false, bLhi.v,
                                                (short)0, sL, false, false);
    sR = __builtin_amdgcn_wmma_f32_16x16x32_f16(false, a_lo.v, false, bRlo.v,
                                                (short)0, sR, false, false);
    sR = __builtin_amdgcn_wmma_f32_16x16x32_f16(false, a_hi.v, false, bRhi.v,
                                                (short)0, sR, false, false);

    if (domask) {
#pragma unroll
      for (int v = 0; v < 8; ++v) {
        const int qrow_idx = q0 + v + 8 * hi;          // C-layout row
        if (k0 + n      > qrow_idx) sL[v] = -3.0e38f;  // C-layout col
        if (k0 + 16 + n > qrow_idx) sR[v] = -3.0e38f;
      }
    }

    // ---- online softmax: row-max via 16-lane shuffle tree; exp in VALU
    float al[8];
#pragma unroll
    for (int v = 0; v < 8; ++v) {
      float rm = hmax16(fmaxf(sL[v], sR[v]));
      float nm = fmaxf(mrow[v], rm);
      al[v] = __expf(mrow[v] - nm);
      mrow[v] = nm;
      float pl = __expf(sL[v] - nm);
      float pr = __expf(sR[v] - nm);
      o0[v] *= al[v]; o1[v] *= al[v]; o2[v] *= al[v]; o3[v] *= al[v];
      // stage P (plain [m][k] layout) for re-layout into the A-matrix format
      pbuf[wave][v + 8 * hi][n]      = (_Float16)pl;
      pbuf[wave][v + 8 * hi][16 + n] = (_Float16)pr;
    }

    // ---- reload P as A-matrix (16x32, contiguous k = 16hi + s); wave-private LDS,
    //      DS ops are in-order within a wave (compiler emits s_wait_dscnt).
    AB pa;
    pa.q[0] = *(const u32x4*)&pbuf[wave][n][16 * hi];     // k = 16hi..16hi+7
    pa.q[1] = *(const u32x4*)&pbuf[wave][n][16 * hi + 8]; // k = 16hi+8..16hi+15

    // ---- row-sum on the tensor core: sums[v] = sum_k P[row v+8hi][k]
    v8f sums = {};
    sums = __builtin_amdgcn_wmma_f32_16x16x32_f16(false, pa.v, false, onesb.v,
                                                  (short)0, sums, false, false);
#pragma unroll
    for (int v = 0; v < 8; ++v) lrow[v] = lrow[v] * al[v] + sums[v];

    // ---- O += P·V : B-matrix from vt rows (d fixed, keys contiguous)
#pragma unroll
    for (int t = 0; t < 4; ++t) {
      const _Float16* vr = vbase + (size_t)(16 * t + n) * SEQ + k0;
      AB vb;
      vb.q[0] = *(const u32x4*)(vr + 16 * hi);
      vb.q[1] = *(const u32x4*)(vr + 16 * hi + 8);
      v8f* ot = (t == 0) ? &o0 : (t == 1) ? &o1 : (t == 2) ? &o2 : &o3;
      *ot = __builtin_amdgcn_wmma_f32_16x16x32_f16(false, pa.v, false, vb.v,
                                                   (short)0, *ot, false, false);
    }
  };

  // Main loop: mask-free tiles strictly below the diagonal tile.
  for (int j = 0; j < nT - 1; ++j) tile_step(j << 5, false);
  // Peeled diagonal tile with causal masking (compile-time specialization).
  tile_step((nT - 1) << 5, true);

  // ---- epilogue: divide by softmax denominator, write fp32 output
  float* obase = out + ((size_t)bh * SEQ + q0) * DHEAD + n;
#pragma unroll
  for (int v = 0; v < 8; ++v) {
    const float inv = 1.0f / lrow[v];
    float* orow = obase + (size_t)(v + 8 * hi) * DHEAD;
    orow[0]  = o0[v] * inv;
    orow[16] = o1[v] * inv;
    orow[32] = o2[v] * inv;
    orow[48] = o3[v] * inv;
  }
}

// ---------------------------------------------------------------------------
// Launch
// ---------------------------------------------------------------------------
extern "C" void kernel_launch(void* const* d_in, const int* in_sizes, int n_in,
                              void* d_out, int out_size, void* d_ws, size_t ws_size,
                              hipStream_t stream) {
  const float* Q    = (const float*)d_in[0];
  const float* K    = (const float*)d_in[1];
  const float* V    = (const float*)d_in[2];
  const float* gain = (const float*)d_in[3];

  const size_t N = (size_t)BH * SEQ * DHEAD;   // elements per tensor
  _Float16* qn = (_Float16*)d_ws;              // 16 MB
  _Float16* kn = qn + N;                       // 16 MB
  _Float16* vt = kn + N;                       // 16 MB  (total 48 MB scratch)

  const int rows   = BH * SEQ;                 // 131072 rows of 64
  const int rblk   = rows / 8;                 // 8 rows (waves) per block
  const float qscl = 0.125f;                   // 1/sqrt(D) folded into qn

  rmsnorm_f32_to_f16<<<rblk, 256, 0, stream>>>(Q, qn, gain, qscl);
  rmsnorm_f32_to_f16<<<rblk, 256, 0, stream>>>(K, kn, nullptr, 1.0f);
  v_f32_to_f16_T<<<dim3(SEQ / 64, BH), 256, 0, stream>>>(V, vt);
  attn_flash_wmma<<<dim3(SEQ / (16 * WAVES), BH), 256, 0, stream>>>(
      qn, kn, vt, (float*)d_out);
}
